// STEGOLoss_57475252355582
// MI455X (gfx1250) — matrix operand order repaired and checked by reading, not compiled, not run on hardware
//
#include <hip/hip_runtime.h>

typedef __attribute__((ext_vector_type(16))) _Float16 v16h;
typedef __attribute__((ext_vector_type(8)))  _Float16 v8h;
typedef __attribute__((ext_vector_type(8)))  float    v8f;

#define DCH   128      // channel dim (K of the dots)
#define NPIX  4096     // H*W
#define INV_TEMP 10.0f // 1 / 0.1

// ---------------------------------------------------------------------------
// Kernel 1: L2-normalize over channels, emit f16 table laid out (B*N, D).
// Reads of (B,D,N) are coalesced across threads (thread = pixel).
// ---------------------------------------------------------------------------
__global__ __launch_bounds__(256) void stego_normalize(
    const float* __restrict__ feats, _Float16* __restrict__ outF, int BN)
{
  int p = blockIdx.x * blockDim.x + threadIdx.x;
  if (p >= BN) return;
  int b = p / NPIX, n = p % NPIX;
  const float* src = feats + (size_t)b * DCH * NPIX + n;

  float ssq = 0.f;
#pragma unroll 8
  for (int d = 0; d < DCH; ++d) {
    float v = src[(size_t)d * NPIX];
    ssq = fmaf(v, v, ssq);
  }
  float scale = 1.0f / fmaxf(sqrtf(ssq), 1e-12f);

  _Float16* dst = outF + (size_t)p * DCH;
#pragma unroll 8
  for (int d = 0; d < DCH; ++d) {
    dst[d] = (_Float16)(src[(size_t)d * NPIX] * scale);  // L2 hit on re-read
  }
}

// ---------------------------------------------------------------------------
// Kernel 2: one wave handles 16 (pos or neg) pairs.
//   A frag: 16 gathered f_i rows in the 16-bit 16x32 A layout
//           (lane -> row L&15; half L>>4 selects K sub-ranges).
//   B frag: 16 gathered f_j rows as a 32x16 column matrix
//           (lane -> col L&15; half selects K 0-15 / 16-31).
//   4 chained v_wmma_f32_16x16x32_f16 accumulate K=128; diagonal of the
//   16x16 result = the 16 desired dot products.
// ---------------------------------------------------------------------------
__global__ __launch_bounds__(256) void stego_pair_wmma(
    const _Float16* __restrict__ F,
    const int* __restrict__ pb, const int* __restrict__ pi, const int* __restrict__ pj,
    const int* __restrict__ nb, const int* __restrict__ ni, const int* __restrict__ nj,
    float* __restrict__ partials, int groupsPos, int totalGroups)
{
  int wave = (int)((blockIdx.x * blockDim.x + threadIdx.x) >> 5);
  int lane = (int)(threadIdx.x & 31);
  if (wave >= totalGroups) return;          // wave-uniform: EXEC stays all-ones

  bool isNeg = (wave >= groupsPos);
  int  g     = isNeg ? wave - groupsPos : wave;
  const int* ib = isNeg ? nb : pb;
  const int* ii = isNeg ? ni : pi;
  const int* ij = isNeg ? nj : pj;

  int r    = lane & 15;
  int half = lane >> 4;
  int idx  = g * 16 + r;
  int bb   = ib[idx];
  const _Float16* rowA = F + ((size_t)bb * NPIX + ii[idx]) * DCH;
  const _Float16* rowB = F + ((size_t)bb * NPIX + ij[idx]) * DCH;

  v8f c = {};
  union { v16h v; v8h h[2]; } a;
#pragma unroll
  for (int kk = 0; kk < DCH; kk += 32) {
    // A 16-bit 16x32 layout: elem e -> K = kk + half*16 + e (matches B)
    a.h[0] = *(const v8h*)(rowA + kk + half * 16);       // 16B aligned
    a.h[1] = *(const v8h*)(rowA + kk + half * 16 + 8);   // 16B aligned
    // B 32x16 layout: elem e -> K = kk + half*16 + e, col = lane&15
    v16h bf = *(const v16h*)(rowB + kk + half * 16);     // 32B aligned
    c = __builtin_amdgcn_wmma_f32_16x16x32_f16(
        false, a.v, false, bf, (short)0, c, false, false);
  }

  // Diagonal of 16x16 f32 C: pair r in 0..7 -> lane r, elem r;
  //                          pair r in 8..15 -> lane r+16, elem r-8.
  float term = 0.f;
  if ((lane < 8) || (lane >= 16 && lane < 24)) {
    int e = lane & 7;
    float dot = 0.f;
#pragma unroll
    for (int k = 0; k < 8; ++k) dot = (k == e) ? c[k] : dot;   // const-index selects
    float s = dot * INV_TEMP;
    float x = isNeg ? s : -s;                 // loss term = softplus(x) = -log(sigmoid(-x))
    term = fmaxf(x, 0.f) + log1pf(__expf(-fabsf(x)));
  }
#pragma unroll
  for (int off = 16; off > 0; off >>= 1)       // wave32 reduction
    term += __shfl_down(term, off, 32);
  if (lane == 0) partials[wave] = term;
}

// ---------------------------------------------------------------------------
// Kernel 3: deterministic single-block reduction (no float atomics).
// ---------------------------------------------------------------------------
__global__ __launch_bounds__(256) void stego_reduce(
    const float* __restrict__ partials, int n, float invP, float* __restrict__ out)
{
  __shared__ float sm[256];
  float acc = 0.f;
  for (int i = (int)threadIdx.x; i < n; i += 256) acc += partials[i];
  sm[threadIdx.x] = acc;
  __syncthreads();
  for (int off = 128; off > 0; off >>= 1) {
    if ((int)threadIdx.x < off) sm[threadIdx.x] += sm[threadIdx.x + off];
    __syncthreads();
  }
  if (threadIdx.x == 0) out[0] = sm[0] * invP;
}

// ---------------------------------------------------------------------------
extern "C" void kernel_launch(void* const* d_in, const int* in_sizes, int n_in,
                              void* d_out, int out_size, void* d_ws, size_t ws_size,
                              hipStream_t stream)
{
  (void)n_in; (void)out_size; (void)ws_size;
  const float* feats = (const float*)d_in[0];
  const int* pb = (const int*)d_in[1];
  const int* pi = (const int*)d_in[2];
  const int* pj = (const int*)d_in[3];
  const int* nb = (const int*)d_in[4];
  const int* ni = (const int*)d_in[5];
  const int* nj = (const int*)d_in[6];

  const int BN = in_sizes[0] / DCH;          // B * H * W = 32768
  const int P  = in_sizes[1];                // 65536 pairs per side
  const int groupsPos   = P / 16;            // 4096
  const int totalGroups = 2 * groupsPos;     // pos + neg waves

  // Workspace layout: [ f16 normalized table (BN*DCH) | float partials ]
  _Float16* F = (_Float16*)d_ws;
  float* partials = (float*)((char*)d_ws + (size_t)BN * DCH * sizeof(_Float16));

  stego_normalize<<<(BN + 255) / 256, 256, 0, stream>>>(feats, F, BN);

  int threadsB = totalGroups * 32;           // one wave per 16-pair group
  stego_pair_wmma<<<(threadsB + 255) / 256, 256, 0, stream>>>(
      F, pb, pi, pj, nb, ni, nj, partials, groupsPos, totalGroups);

  stego_reduce<<<1, 256, 0, stream>>>(partials, totalGroups,
                                      1.0f / (float)P, (float*)d_out);
}
